// feature_extractor_85864986182372
// MI455X (gfx1250) — matrix operand order repaired and checked by reading, not compiled, not run on hardware
//
#include <hip/hip_runtime.h>
#include <stdint.h>

// ---- problem constants (Swin window attention) ----
#define TOK    98304   // B_ * N = 1536 * 64 rows
#define DIMC   192
#define HEADS  6
#define HDIM   32
#define NTOK   64
#define NWIN   192     // windows per image (mask dim 0)
#define BWIN   1536    // B_ = 8 * 192
#define QK_SCALE 0.17677669529663689f   // 1/sqrt(32)

typedef __attribute__((ext_vector_type(16))) __bf16 v16bf;
typedef __attribute__((ext_vector_type(8)))  float  v8f;

// 16 bf16 = 32B fragment; uint4 members give guaranteed b128 loads.
union BF16x16 { v16bf v; unsigned short s[16]; uint4 q[2]; };

static __device__ __forceinline__ unsigned short f2bf(float f) {
  uint32_t u = __builtin_bit_cast(uint32_t, f);
  u += 0x7FFFu + ((u >> 16) & 1u);          // round-to-nearest-even
  return (unsigned short)(u >> 16);
}

// =====================================================================
// Kernel 0: one-shot f32 -> bf16 conversion (8 elems / thread, b128 I/O).
// Removes all conversion VALU from the WMMA inner loops.
// =====================================================================
__global__ __launch_bounds__(256)
void cvt_bf16_kernel(const float* __restrict__ src,
                     unsigned short* __restrict__ dst, int n8)
{
  const int i = blockIdx.x * 256 + threadIdx.x;
  if (i >= n8) return;
  const float4* s = (const float4*)src + 2 * (size_t)i;
  const float4 f0 = s[0], f1 = s[1];
  union { unsigned short h[8]; uint4 q; } u;
  u.h[0] = f2bf(f0.x); u.h[1] = f2bf(f0.y); u.h[2] = f2bf(f0.z); u.h[3] = f2bf(f0.w);
  u.h[4] = f2bf(f1.x); u.h[5] = f2bf(f1.y); u.h[6] = f2bf(f1.z); u.h[7] = f2bf(f1.w);
  ((uint4*)dst)[i] = u.q;
}

// =====================================================================
// Kernel 1: fused QKV projection (bf16 in, bf16 out).
//   cols 0..191   : Q = (x @ wq^T + bq) * scale   -> Qbf[b,h,tok,d]
//   cols 192..383 : K                              -> Kbf[b,h,tok,d]
//   cols 384..575 : V                              -> Vtbf[b,h,d,tok] (transposed)
// 64x64 C tile per workgroup; 8 waves, 2 WMMA tiles each; K unrolled (6x32).
// =====================================================================
__global__ __launch_bounds__(256)
void qkv_proj_kernel(const unsigned short* __restrict__ Xbf,
                     const unsigned short* __restrict__ PXbf,
                     const unsigned short* __restrict__ Wqb,
                     const unsigned short* __restrict__ Wkvb,
                     const float* __restrict__ bq, const float* __restrict__ bkv,
                     unsigned short* __restrict__ Qbf,
                     unsigned short* __restrict__ Kbf,
                     unsigned short* __restrict__ Vtbf)
{
  const int nt   = blockIdx.x;          // 0..8  (64-col tile of 576 fused cols)
  const int mt   = blockIdx.y;          // 0..1535
  const int wave = threadIdx.x >> 5;
  const int lane = threadIdx.x & 31;
  const int lh   = lane >> 4;           // lane half (K striping select)
  const int ln   = lane & 15;

  const int m0 = (wave >> 1) * 16;      // 0,16,32,48
  const int n0 = (wave & 1) * 32;       // 0 or 32
  const int rowBase = mt * 64;
  const int colBase = nt * 64;

  const bool is_q = (nt < 3);
  const unsigned short* __restrict__ A = is_q ? Xbf : PXbf;
  const unsigned short* __restrict__ W = is_q ? Wqb : Wkvb;
  const int wcolBase = colBase - (is_q ? 0 : 192);   // row index into W

  const unsigned short* arow  = A + (size_t)(rowBase + m0 + ln) * DIMC;
  const unsigned short* wrow0 = W + (size_t)(wcolBase + n0 + ln) * DIMC;
  const unsigned short* wrow1 = W + (size_t)(wcolBase + n0 + 16 + ln) * DIMC;

  v8f acc0 = {}; v8f acc1 = {};
#pragma unroll
  for (int k = 0; k < DIMC; k += 32) {
    BF16x16 a, b0, b1;
    // A 16x32: elems = K(lh*16 + 0..15)
    a.q[0]  = *(const uint4*)(arow + k + lh*16);
    a.q[1]  = *(const uint4*)(arow + k + lh*16 + 8);
    // B 32x16: lane = col, elems = K(lh*16 + 0..15)
    b0.q[0] = *(const uint4*)(wrow0 + k + lh*16);
    b0.q[1] = *(const uint4*)(wrow0 + k + lh*16 + 8);
    b1.q[0] = *(const uint4*)(wrow1 + k + lh*16);
    b1.q[1] = *(const uint4*)(wrow1 + k + lh*16 + 8);
    acc0 = __builtin_amdgcn_wmma_f32_16x16x32_bf16(false, a.v, false, b0.v, (short)0, acc0, false, false);
    acc1 = __builtin_amdgcn_wmma_f32_16x16x32_bf16(false, a.v, false, b1.v, (short)0, acc1, false, false);
  }

#pragma unroll
  for (int t = 0; t < 2; ++t) {
    v8f acc = t ? acc1 : acc0;
    const int c0 = colBase + n0 + t * 16;
#pragma unroll
    for (int r = 0; r < 8; ++r) {            // C layout: M = r + 8*lh, N = ln
      const int gr  = rowBase + m0 + r + lh * 8;
      const int tok = gr & 63;
      const int bw  = gr >> 6;
      const int gc  = c0 + ln;
      float v = acc[r];
      if (gc < 192) {
        const int h = gc >> 5, d = gc & 31;
        v = (v + bq[gc]) * QK_SCALE;
        Qbf[((size_t)(bw * HEADS + h) * NTOK + tok) * HDIM + d] = f2bf(v);
      } else {
        const int f2 = gc - 192;
        v += bkv[f2];
        const int h = (f2 % 192) >> 5, d = f2 & 31;
        if (f2 < 192)
          Kbf[((size_t)(bw * HEADS + h) * NTOK + tok) * HDIM + d] = f2bf(v);
        else
          Vtbf[((size_t)(bw * HEADS + h) * HDIM + d) * NTOK + tok] = f2bf(v);
      }
    }
  }
}

// =====================================================================
// Kernel 2: fused attention core per (window, head).
//   S = Q K^T (+bias +mask) -> softmax -> O = P V.   4 waves, 16 rows each.
// =====================================================================
__global__ __launch_bounds__(128)
void window_attn_kernel(const unsigned short* __restrict__ Qbf,
                        const unsigned short* __restrict__ Kbf,
                        const unsigned short* __restrict__ Vtbf,
                        const float* __restrict__ mask,
                        const int*   __restrict__ rpi,
                        const float* __restrict__ rpb,
                        unsigned short* __restrict__ AO)
{
  const int bh   = blockIdx.x;            // 0..9215
  const int b    = bh / HEADS;
  const int h    = bh % HEADS;
  const int wv   = threadIdx.x >> 5;      // 0..3
  const int lane = threadIdx.x & 31;
  const int lh   = lane >> 4, ln = lane & 15;
  const int m0   = wv * 16;

  const unsigned short* Qh = Qbf  + (size_t)bh * NTOK * HDIM;
  const unsigned short* Kh = Kbf  + (size_t)bh * NTOK * HDIM;
  const unsigned short* Vh = Vtbf + (size_t)bh * HDIM * NTOK;
  const float* mw = mask + (size_t)(b % NWIN) * NTOK * NTOK;

  // warm the V^T rows this lane will consume after softmax (global_prefetch_b8)
  __builtin_prefetch(Vh + ln * NTOK, 0, 3);
  __builtin_prefetch(Vh + (16 + ln) * NTOK, 0, 3);

  __shared__ __align__(16) unsigned short Plds[4][16][64];

  // ---- S = Q K^T : one WMMA per 16x16 tile (K = d = 32) ----
  BF16x16 aQ;
  {
    const unsigned short* qr = Qh + (m0 + ln) * HDIM;
    aQ.q[0] = *(const uint4*)(qr + lh*16);
    aQ.q[1] = *(const uint4*)(qr + lh*16 + 8);
  }
  v8f S[4];
#pragma unroll
  for (int t = 0; t < 4; ++t) {
    BF16x16 bK;
    const unsigned short* kr = Kh + (t*16 + ln) * HDIM;   // B col = key token
    bK.q[0] = *(const uint4*)(kr + lh*16);
    bK.q[1] = *(const uint4*)(kr + lh*16 + 8);
    v8f z = {};
    S[t] = __builtin_amdgcn_wmma_f32_16x16x32_bf16(false, aQ.v, false, bK.v, (short)0, z, false, false);
  }

  // ---- relative-position bias + shift mask (L2-resident tables) ----
#pragma unroll
  for (int t = 0; t < 4; ++t) {
#pragma unroll
    for (int r = 0; r < 8; ++r) {
      const int i = m0 + r + lh * 8;
      const int j = t * 16 + ln;
      S[t][r] += rpb[rpi[i * NTOK + j] * HEADS + h] + mw[i * NTOK + j];
    }
  }

  // ---- row softmax: row (m0 + r + 8*lh) lives in one 16-lane half ----
  float rinv[8];
#pragma unroll
  for (int r = 0; r < 8; ++r) {
    float m = fmaxf(fmaxf(S[0][r], S[1][r]), fmaxf(S[2][r], S[3][r]));
#pragma unroll
    for (int d = 8; d >= 1; d >>= 1) m = fmaxf(m, __shfl_xor(m, d, 16));
    float sum = 0.f;
#pragma unroll
    for (int t = 0; t < 4; ++t) {
      float e = __expf(S[t][r] - m);
      S[t][r] = e;
      sum += e;
    }
#pragma unroll
    for (int d = 8; d >= 1; d >>= 1) sum += __shfl_xor(sum, d, 16);
    rinv[r] = 1.0f / sum;
  }

  // ---- P to LDS (C layout -> row-major bf16 strip) ----
#pragma unroll
  for (int t = 0; t < 4; ++t)
#pragma unroll
    for (int r = 0; r < 8; ++r)
      Plds[wv][r + lh * 8][t * 16 + ln] = f2bf(S[t][r] * rinv[r]);

  __syncthreads();

  // ---- O = P V : 2 d-tiles x 2 token-chunks of 32 ----
#pragma unroll
  for (int dt = 0; dt < 2; ++dt) {
    v8f O = {};
#pragma unroll
    for (int kc = 0; kc < 2; ++kc) {
      BF16x16 aP, bV;
      aP.q[0] = *(const uint4*)&Plds[wv][ln][kc*32 + lh*16];
      aP.q[1] = *(const uint4*)&Plds[wv][ln][kc*32 + lh*16 + 8];
      const unsigned short* vr = Vh + (dt*16 + ln) * NTOK;  // V^T row = dim, contiguous tokens
      bV.q[0] = *(const uint4*)(vr + kc*32 + lh*16);
      bV.q[1] = *(const uint4*)(vr + kc*32 + lh*16 + 8);
      O = __builtin_amdgcn_wmma_f32_16x16x32_bf16(false, aP.v, false, bV.v, (short)0, O, false, false);
    }
#pragma unroll
    for (int r = 0; r < 8; ++r) {
      const int tok = m0 + r + lh * 8;
      const int d   = dt * 16 + ln;
      AO[((size_t)b * NTOK + tok) * DIMC + h * HDIM + d] = f2bf(O[r]);
    }
  }
}

// =====================================================================
// Kernel 3: output projection  out = AO(bf16) @ wp^T + bp  (f32 out)
// =====================================================================
__global__ __launch_bounds__(256)
void out_proj_kernel(const unsigned short* __restrict__ AO,
                     const unsigned short* __restrict__ Wpb,
                     const float* __restrict__ bp,
                     float* __restrict__ out)
{
  const int nt   = blockIdx.x;           // 0..2
  const int mt   = blockIdx.y;           // 0..1535
  const int wave = threadIdx.x >> 5;
  const int lane = threadIdx.x & 31;
  const int lh   = lane >> 4, ln = lane & 15;
  const int m0 = (wave >> 1) * 16;
  const int n0 = (wave & 1) * 32;
  const int rowBase = mt * 64, colBase = nt * 64;

  const unsigned short* arow  = AO  + (size_t)(rowBase + m0 + ln) * DIMC;
  const unsigned short* wrow0 = Wpb + (size_t)(colBase + n0 + ln) * DIMC;
  const unsigned short* wrow1 = Wpb + (size_t)(colBase + n0 + 16 + ln) * DIMC;

  v8f acc0 = {}, acc1 = {};
#pragma unroll
  for (int k = 0; k < DIMC; k += 32) {
    BF16x16 a, b0, b1;
    a.q[0]  = *(const uint4*)(arow + k + lh*16);
    a.q[1]  = *(const uint4*)(arow + k + lh*16 + 8);
    b0.q[0] = *(const uint4*)(wrow0 + k + lh*16);
    b0.q[1] = *(const uint4*)(wrow0 + k + lh*16 + 8);
    b1.q[0] = *(const uint4*)(wrow1 + k + lh*16);
    b1.q[1] = *(const uint4*)(wrow1 + k + lh*16 + 8);
    acc0 = __builtin_amdgcn_wmma_f32_16x16x32_bf16(false, a.v, false, b0.v, (short)0, acc0, false, false);
    acc1 = __builtin_amdgcn_wmma_f32_16x16x32_bf16(false, a.v, false, b1.v, (short)0, acc1, false, false);
  }

#pragma unroll
  for (int t = 0; t < 2; ++t) {
    v8f acc = t ? acc1 : acc0;
    const int gc = colBase + n0 + t * 16 + ln;
#pragma unroll
    for (int r = 0; r < 8; ++r) {
      const int gr = rowBase + m0 + r + lh * 8;
      out[(size_t)gr * DIMC + gc] = acc[r] + bp[gc];
    }
  }
}

// =====================================================================
extern "C" void kernel_launch(void* const* d_in, const int* in_sizes, int n_in,
                              void* d_out, int out_size, void* d_ws, size_t ws_size,
                              hipStream_t stream)
{
  (void)in_sizes; (void)n_in; (void)out_size; (void)ws_size;
  const float* x    = (const float*)d_in[0];
  const float* px   = (const float*)d_in[1];
  const float* mask = (const float*)d_in[2];
  const int*   rpi  = (const int*)  d_in[3];
  const float* rpb  = (const float*)d_in[4];
  const float* wq   = (const float*)d_in[5];
  const float* bq   = (const float*)d_in[6];
  const float* wkv  = (const float*)d_in[7];
  const float* bkv  = (const float*)d_in[8];
  const float* wp   = (const float*)d_in[9];
  const float* bp   = (const float*)d_in[10];
  float* out = (float*)d_out;

  const size_t elems = (size_t)TOK * DIMC;               // 18,874,368
  unsigned short* Qbf  = (unsigned short*)d_ws;          // bf16 workspaces
  unsigned short* Kbf  = Qbf + elems;
  unsigned short* Vt   = Kbf + elems;
  unsigned short* AO   = Vt  + elems;
  unsigned short* Xbf  = AO  + elems;
  unsigned short* PXbf = Xbf + elems;
  unsigned short* Wqb  = PXbf + elems;                   // 192*192
  unsigned short* Wkvb = Wqb + 192*192;                  // 384*192
  unsigned short* Wpb  = Wkvb + 384*192;                 // 192*192  (total ~227 MB)

  const int n8x = (int)(elems / 8);
  cvt_bf16_kernel<<<(n8x + 255) / 256, 256, 0, stream>>>(x,   Xbf,  n8x);
  cvt_bf16_kernel<<<(n8x + 255) / 256, 256, 0, stream>>>(px,  PXbf, n8x);
  cvt_bf16_kernel<<<(192*192/8 + 255) / 256, 256, 0, stream>>>(wq,  Wqb,  192*192/8);
  cvt_bf16_kernel<<<(384*192/8 + 255) / 256, 256, 0, stream>>>(wkv, Wkvb, 384*192/8);
  cvt_bf16_kernel<<<(192*192/8 + 255) / 256, 256, 0, stream>>>(wp,  Wpb,  192*192/8);

  qkv_proj_kernel<<<dim3(9, BWIN), 256, 0, stream>>>(Xbf, PXbf, Wqb, Wkvb, bq, bkv, Qbf, Kbf, Vt);
  window_attn_kernel<<<dim3(BWIN * HEADS), 128, 0, stream>>>(Qbf, Kbf, Vt, mask, rpi, rpb, AO);
  out_proj_kernel<<<dim3(3, BWIN), 256, 0, stream>>>(AO, wp ? Wpb : Wpb, bp, out);
}
